// MultiHeadSelfAttention_76914274337300
// MI455X (gfx1250) — matrix operand order repaired and checked by reading, not compiled, not run on hardware
//
#include <hip/hip_runtime.h>
#include <hip/hip_bf16.h>

// ---------------------------------------------------------------------------
// Fused MHA for MI455X (gfx1250, wave32, WMMA bf16).
// GEMM operands pre-packed into fragment-major bf16 tiles (aligned 32B loads);
// flash attention with 32 q-rows/wave, WMMA row-sum softmax, and direct
// A-fragment-packed output.
// ---------------------------------------------------------------------------

typedef __bf16 bf16_t;
typedef __bf16 v16bf __attribute__((ext_vector_type(16)));
typedef float  v8f   __attribute__((ext_vector_type(8)));

constexpr int B_  = 2;
constexpr int T_  = 2048;
constexpr int C_  = 1024;
constexpr int H_  = 16;
constexpr int DH  = 64;
constexpr int BT  = B_ * T_;     // 4096
constexpr int N3  = 3 * C_;      // 3072
constexpr int KT_C = C_ / 32;    // 32 K-tiles for K=1024

__device__ __forceinline__ v8f wmma_bf16(v16bf a, v16bf b, v8f c) {
  return __builtin_amdgcn_wmma_f32_16x16x32_bf16(false, a, false, b,
                                                 (short)0, c, false, false);
}

// A-matrix 16x32 (MxK): K-column for vgpr j, lane-half h (ISA 7.12.2)
__device__ __forceinline__ int acol(int j, int half) {
  return ((j & 4) << 2) + half * 8 + ((j & 3) << 1);
}

// B fragment 32x16 (KxN) from source stored row-major as N x K (B^T view):
// lane&15 = N row, K = 8*(lane>>4) + e (e<8), 16 + 8*(lane>>4) + (e-8) (e>=8).
__device__ __forceinline__ v16bf load_fragB_nk(const bf16_t* src, int ld) {
  typedef __bf16 v8bf __attribute__((ext_vector_type(8)));
  const int lane = threadIdx.x & 31;
  const bf16_t* p = src + (lane & 15) * ld + (lane >> 4) * 8;
  const v8bf lo = *(const v8bf*)p;
  const v8bf hi = *(const v8bf*)(p + 16);
  return __builtin_shufflevector(lo, hi, 0, 1, 2, 3, 4, 5, 6, 7, 8, 9, 10, 11, 12, 13, 14, 15);
}

// ---------------------------------------------------------------------------
// Packing kernels: one thread per packed u32 (2 bf16).
// A-pack layout: tile = mt*KT + kt, 512 bf16 per tile, elem = lane*16 + 2j+s.
// ---------------------------------------------------------------------------
__global__ __launch_bounds__(256) void pack_a_f32(const float* __restrict__ src,
                                                  unsigned* __restrict__ dst,
                                                  int K) {
  const int KT = K >> 5;
  const int idx = blockIdx.x * 256 + threadIdx.x;
  const int p = idx & 255, tile = idx >> 8;
  const int kt = tile % KT, mt = tile / KT;
  const int j = p & 7, lane = p >> 3, r = lane & 15, half = lane >> 4;
  const int c = acol(j, half);
  const float* s = src + (size_t)(mt * 16 + r) * K + kt * 32 + c;
  union { bf16_t h[2]; unsigned u; } u;
  u.h[0] = (bf16_t)s[0];
  u.h[1] = (bf16_t)s[1];
  dst[idx] = u.u;
}

// W-pack: tile = nt*KT + kt; lane&15 = N col, K = acol(j, lane>>4) + s.
__global__ __launch_bounds__(256) void pack_w(const float* __restrict__ src,
                                              unsigned* __restrict__ dst,
                                              int K, int N) {
  const int KT = K >> 5;
  const int idx = blockIdx.x * 256 + threadIdx.x;
  const int p = idx & 255, tile = idx >> 8;
  const int kt = tile % KT, nt = tile / KT;
  const int j = p & 7, lane = p >> 3, n = lane & 15, half = lane >> 4;
  const int k0 = acol(j, half);
  const float* s = src + (size_t)(kt * 32 + k0) * N + nt * 16 + n;
  union { bf16_t h[2]; unsigned u; } u;
  u.h[0] = (bf16_t)s[0];
  u.h[1] = (bf16_t)s[N];
  dst[idx] = u.u;
}

// ---------------------------------------------------------------------------
// Packed GEMMs: wave computes a 32x64 tile (2 A-frags x 4 B-frags reused),
// 8 WMMA per K-step against 6 v16bf loads.
// ---------------------------------------------------------------------------
__global__ __launch_bounds__(128) void gemm_qkv(const v16bf* __restrict__ ap,
                                                const v16bf* __restrict__ wp,
                                                bf16_t* __restrict__ out) {
  const int wave = blockIdx.x * 4 + (threadIdx.x >> 5);
  const int NB = N3 / 64;                       // 48
  const int mt2 = wave / NB, nb = wave % NB;
  const int lane = threadIdx.x & 31;
  const v16bf* a0 = ap + (size_t)(mt2 * 2 + 0) * KT_C * 32 + lane;
  const v16bf* a1 = ap + (size_t)(mt2 * 2 + 1) * KT_C * 32 + lane;
  const v16bf* bb = wp + (size_t)(nb * 4) * KT_C * 32 + lane;
  v8f acc[2][4] = {};
  for (int kt = 0; kt < KT_C; ++kt) {
    const v16bf A0 = a0[kt * 32];
    const v16bf A1 = a1[kt * 32];
    __builtin_prefetch(a0 + (kt + 2) * 32, 0, 1);
    __builtin_prefetch(bb + (kt + 2) * 32, 0, 1);
#pragma unroll
    for (int nn = 0; nn < 4; ++nn) {
      const v16bf Bn = bb[(nn * KT_C + kt) * 32];
      acc[0][nn] = wmma_bf16(A0, Bn, acc[0][nn]);
      acc[1][nn] = wmma_bf16(A1, Bn, acc[1][nn]);
    }
  }
  const int half = lane >> 4, n = lane & 15;
#pragma unroll
  for (int s = 0; s < 2; ++s)
#pragma unroll
    for (int nn = 0; nn < 4; ++nn)
#pragma unroll
      for (int i = 0; i < 8; ++i)
        out[(size_t)(mt2 * 32 + s * 16 + i + 8 * half) * N3 +
            nb * 64 + nn * 16 + n] = (bf16_t)acc[s][nn][i];
}

__global__ __launch_bounds__(128) void gemm_out(const v16bf* __restrict__ ap,
                                                const v16bf* __restrict__ wp,
                                                float* __restrict__ out) {
  const int wave = blockIdx.x * 4 + (threadIdx.x >> 5);
  const int NB = C_ / 64;                       // 16
  const int mt2 = wave / NB, nb = wave % NB;
  const int lane = threadIdx.x & 31;
  const v16bf* a0 = ap + (size_t)(mt2 * 2 + 0) * KT_C * 32 + lane;
  const v16bf* a1 = ap + (size_t)(mt2 * 2 + 1) * KT_C * 32 + lane;
  const v16bf* bb = wp + (size_t)(nb * 4) * KT_C * 32 + lane;
  v8f acc[2][4] = {};
  for (int kt = 0; kt < KT_C; ++kt) {
    const v16bf A0 = a0[kt * 32];
    const v16bf A1 = a1[kt * 32];
    __builtin_prefetch(a0 + (kt + 2) * 32, 0, 1);
    __builtin_prefetch(bb + (kt + 2) * 32, 0, 1);
#pragma unroll
    for (int nn = 0; nn < 4; ++nn) {
      const v16bf Bn = bb[(nn * KT_C + kt) * 32];
      acc[0][nn] = wmma_bf16(A0, Bn, acc[0][nn]);
      acc[1][nn] = wmma_bf16(A1, Bn, acc[1][nn]);
    }
  }
  const int half = lane >> 4, n = lane & 15;
#pragma unroll
  for (int s = 0; s < 2; ++s)
#pragma unroll
    for (int nn = 0; nn < 4; ++nn)
#pragma unroll
      for (int i = 0; i < 8; ++i)
        out[(size_t)(mt2 * 32 + s * 16 + i + 8 * half) * C_ +
            nb * 64 + nn * 16 + n] = acc[s][nn][i];
}

// ---------------------------------------------------------------------------
// RoPE: reads row-major qkv, writes
//   q -> A-fragment-packed tiles [b][h][qt][frag0|frag1][512]  (scaled 1/8)
//   k -> row-major [B,H,T,Dh]
//   v -> transposed [B,H,Dh,T]
// ---------------------------------------------------------------------------
__global__ __launch_bounds__(256) void rope_kernel(const bf16_t* __restrict__ qkv,
                                                   bf16_t* __restrict__ qpack,
                                                   bf16_t* __restrict__ k,
                                                   bf16_t* __restrict__ v) {
  const int idx = blockIdx.x * 256 + threadIdx.x;
  const int d = idx & 31;
  const int t = (idx >> 5) & (T_ - 1);
  const int h = (idx >> 16) & (H_ - 1);
  const int b = idx >> 20;

  const bf16_t* row = qkv + (size_t)(b * T_ + t) * N3 + h * DH;
  const float ang = (float)t * __expf(-(float)d * 0.28782313662425572f);
  float sn, cs;
  __sincosf(ang, &sn, &cs);

  const float q0 = (float)row[d],          q1 = (float)row[d + 32];
  const float qe = (float)row[2 * d],      qo = (float)row[2 * d + 1];
  const float k0 = (float)row[C_ + d],     k1 = (float)row[C_ + d + 32];
  const float ke = (float)row[C_ + 2 * d], ko = (float)row[C_ + 2 * d + 1];
  const float v0 = (float)row[2 * C_ + d], v1 = (float)row[2 * C_ + d + 32];

  const int r = t & 15;
  const int l1 = r + 16 * ((d >> 3) & 1);
  const int e1 = ((d >> 4) << 3) + (d & 7);
  const size_t tb = ((size_t)(b * H_ + h) * (T_ / 16) + (t >> 4)) * 1024;
  qpack[tb + l1 * 16 + e1]       = (bf16_t)((q0 * cs - qo * sn) * 0.125f);
  qpack[tb + 512 + l1 * 16 + e1] = (bf16_t)((q1 * cs + qe * sn) * 0.125f);

  const size_t kbase = (size_t)((b * H_ + h) * T_ + t) * DH;
  k[kbase + d]      = (bf16_t)(k0 * cs - ko * sn);
  k[kbase + d + 32] = (bf16_t)(k1 * cs + ke * sn);
  const size_t vbase = (size_t)((b * H_ + h) * DH + d) * T_ + t;
  v[vbase]           = (bf16_t)v0;
  v[vbase + 32 * T_] = (bf16_t)v1;
}

// ---------------------------------------------------------------------------
// Flash attention: one wave owns 32 q-rows (two 16-row subtiles sharing the
// K and V fragments).  Causal tile skipping per-subtile is wave-uniform so
// EXEC stays all-ones for WMMA.  Row sums via WMMA against an all-ones B
// fragment (replaces shuffle reduction; l matches the bf16 P used for P.V).
// Output written directly in A-fragment-packed layout for the final GEMM.
// ---------------------------------------------------------------------------
__global__ __launch_bounds__(128) void attn_kernel(const v16bf* __restrict__ qp,
                                                   const bf16_t* __restrict__ k,
                                                   const bf16_t* __restrict__ v,
                                                   bf16_t* __restrict__ apk) {
  __shared__ __align__(32) bf16_t ldsP[4][2][512];
  const int wv   = threadIdx.x >> 5;
  const int wave = blockIdx.x * 4 + wv;
  const int qt2 = wave & (T_ / 32 - 1);          // 64 double-tiles
  const int h   = (wave >> 6) & (H_ - 1);
  const int b   = wave >> 10;
  const int bh  = b * H_ + h;
  const int qb[2] = { qt2 * 32, qt2 * 32 + 16 };
  const int lane = threadIdx.x & 31, half = lane >> 4, nidx = lane & 15;

  const bf16_t* kp0 = k + (size_t)bh * T_ * DH;
  const bf16_t* vp0 = v + (size_t)bh * DH * T_;

  v16bf aQ[2][2];
#pragma unroll
  for (int u = 0; u < 2; ++u) {
    const size_t qtb = ((size_t)bh * (T_ / 16) + qt2 * 2 + u) * 64;
    aQ[u][0] = qp[qtb + lane];
    aQ[u][1] = qp[qtb + 32 + lane];
  }

  v16bf onesf;
#pragma unroll
  for (int e = 0; e < 16; ++e) onesf[e] = (bf16_t)1.0f;

  v8f accO[2][4] = {};
  float mrow[2][8], lrow[2][8];
#pragma unroll
  for (int u = 0; u < 2; ++u)
#pragma unroll
    for (int i = 0; i < 8; ++i) { mrow[u][i] = -1e30f; lrow[u][i] = 0.0f; }

  for (int kb = 0; kb <= qb[1] + 15; kb += 32) {
    const bool do0 = (kb <= qb[0] + 15);         // wave-uniform
    const bf16_t* kp = kp0 + (size_t)kb * DH;
    __builtin_prefetch(kp + 32 * DH, 0, 1);
    v16bf kf[4];
    kf[0] = load_fragB_nk(kp, DH);
    kf[1] = load_fragB_nk(kp + 32, DH);
    kf[2] = load_fragB_nk(kp + 16 * DH, DH);
    kf[3] = load_fragB_nk(kp + 16 * DH + 32, DH);

#pragma unroll
    for (int u = 0; u < 2; ++u) {
      if (u == 0 && !do0) continue;
      v8f s0 = {}, s1 = {};
      s0 = wmma_bf16(aQ[u][0], kf[0], s0);
      s0 = wmma_bf16(aQ[u][1], kf[1], s0);
      s1 = wmma_bf16(aQ[u][0], kf[2], s1);
      s1 = wmma_bf16(aQ[u][1], kf[3], s1);
#pragma unroll
      for (int i = 0; i < 8; ++i) {
        const int qg = qb[u] + i + 8 * half;
        float x0 = (kb + nidx <= qg)      ? s0[i] : -1e30f;
        float x1 = (kb + 16 + nidx <= qg) ? s1[i] : -1e30f;
        float rmx = fmaxf(x0, x1);
#pragma unroll
        for (int dx = 1; dx < 16; dx <<= 1) rmx = fmaxf(rmx, __shfl_xor(rmx, dx));
        const float mnew  = fmaxf(mrow[u][i], rmx);
        const float alpha = __expf(mrow[u][i] - mnew);
        mrow[u][i] = mnew;
        lrow[u][i] *= alpha;
        const float p0 = __expf(x0 - mnew);
        const float p1 = __expf(x1 - mnew);
#pragma unroll
        for (int nn = 0; nn < 4; ++nn) accO[u][nn][i] *= alpha;
        // store P in A-fragment order: c=n -> elem n&7; c=16+n -> elem +8
        const int l2 = (i + 8 * half) + ((nidx & 8) << 1);
        bf16_t* dstp = &ldsP[wv][u][l2 * 16 + (nidx & 7)];
        dstp[0] = (bf16_t)p0;
        dstp[8] = (bf16_t)p1;
      }
    }
    // same-wave cross-lane LDS RAW
    asm volatile("s_wait_dscnt 0" ::: "memory");

    const bf16_t* vt = vp0 + kb;
    __builtin_prefetch(vt + 32, 0, 1);
    v16bf vf[4];
#pragma unroll
    for (int nn = 0; nn < 4; ++nn)
      vf[nn] = load_fragB_nk(vt + (size_t)nn * 16 * T_, T_);

#pragma unroll
    for (int u = 0; u < 2; ++u) {
      if (u == 0 && !do0) continue;
      const v16bf pF = *(const v16bf*)(&ldsP[wv][u][lane * 16]);
      v8f rs = {};
      rs = wmma_bf16(pF, onesf, rs);        // row sums, replicated per lane
#pragma unroll
      for (int i = 0; i < 8; ++i) lrow[u][i] += rs[i];
#pragma unroll
      for (int nn = 0; nn < 4; ++nn)
        accO[u][nn] = wmma_bf16(pF, vf[nn], accO[u][nn]);
    }
  }

  // epilogue: write directly in A-fragment-packed layout (K = C_ = 1024)
#pragma unroll
  for (int u = 0; u < 2; ++u) {
    const int rowt = (b * T_ + qb[u]) >> 4;        // 16-row tile index
#pragma unroll
    for (int i = 0; i < 8; ++i) {
      const int m = i + 8 * half;
      const float inv = 1.0f / lrow[u][i];
#pragma unroll
      for (int nn = 0; nn < 4; ++nn) {
        const int ct = h * 2 + (nn >> 1);          // 32-col tile index
        const int l2 = m + 16 * (nidx >> 3);
        const int e  = ((nn & 1) << 3) + (nidx & 7);
        apk[((size_t)rowt * KT_C + ct) * 512 + l2 * 16 + e] =
            (bf16_t)(accO[u][nn][i] * inv);
      }
    }
  }
}

// ---------------------------------------------------------------------------
extern "C" void kernel_launch(void* const* d_in, const int* in_sizes, int n_in,
                              void* d_out, int out_size, void* d_ws, size_t ws_size,
                              hipStream_t stream) {
  const float* x     = (const float*)d_in[0];
  const float* w_qkv = (const float*)d_in[1];
  const float* w_out = (const float*)d_in[2];
  float* out = (float*)d_out;

  char* p = (char*)d_ws;
  bf16_t* xpack = (bf16_t*)p; p += (size_t)BT * C_ * 2;   //  8.4 MB
  bf16_t* wqp   = (bf16_t*)p; p += (size_t)C_ * N3 * 2;   //  6.3 MB
  bf16_t* wop   = (bf16_t*)p; p += (size_t)C_ * C_ * 2;   //  2.1 MB
  bf16_t* qkv   = (bf16_t*)p; p += (size_t)BT * N3 * 2;   // 25.2 MB
  bf16_t* qpk   = (bf16_t*)p; p += (size_t)BT * C_ * 2;   //  8.4 MB
  bf16_t* kws   = (bf16_t*)p; p += (size_t)BT * C_ * 2;   //  8.4 MB
  bf16_t* vws   = (bf16_t*)p; p += (size_t)BT * C_ * 2;   //  8.4 MB
  bf16_t* apk   = xpack;  // x dead after gemm_qkv -> attn writes packed here

  pack_a_f32<<<8192, 256, 0, stream>>>(x, (unsigned*)xpack, C_);
  pack_w<<<6144, 256, 0, stream>>>(w_qkv, (unsigned*)wqp, C_, N3);
  pack_w<<<2048, 256, 0, stream>>>(w_out, (unsigned*)wop, C_, C_);

  // qkv projection: (4096/32)*(3072/64) = 6144 waves
  gemm_qkv<<<1536, 128, 0, stream>>>((const v16bf*)xpack, (const v16bf*)wqp, qkv);

  // RoPE + layout: 2^21 threads
  rope_kernel<<<8192, 256, 0, stream>>>(qkv, qpk, kws, vws);

  // flash attention: 2*16*64 = 2048 waves (32 q-rows each)
  attn_kernel<<<512, 128, 0, stream>>>((const v16bf*)qpk, kws, vws, apk);

  // output projection: (4096/32)*(1024/64) = 2048 waves
  gemm_out<<<512, 128, 0, stream>>>((const v16bf*)apk, (const v16bf*)wop, out);
}
